// DeformCenterAttention_67422396612980
// MI455X (gfx1250) — matrix operand order repaired and checked by reading, not compiled, not run on hardware
//
#include <hip/hip_runtime.h>
#include <hip/hip_bf16.h>

// ---- problem constants (from reference) ----
#define Bn  2
#define Hn  64
#define Wn  64
#define Cn  256
#define Gn  8
#define Kn  9
#define CQn 64
#define HCn 32          // == wave32 lanes, one lane per head channel
#define C3n 768         // 3*C
#define Pn  (Hn*Wn)     // 4096
#define Mtok (Bn*Pn)    // 8192 tokens

#define AS1 __attribute__((address_space(1)))
#define AS3 __attribute__((address_space(3)))

typedef __bf16 bf16_t;
typedef __attribute__((ext_vector_type(16))) __bf16 v16bf;
typedef __attribute__((ext_vector_type(8)))  float  v8f;
typedef __attribute__((ext_vector_type(4)))  int    v4i_t;

static __host__ __device__ inline int cdiv_i(int a, int b) { return (a + b - 1) / b; }

// ---------------------------------------------------------------------------
// elementwise fp32 -> bf16
__global__ void k_f32_to_bf16(const float* __restrict__ in, bf16_t* __restrict__ out, int n) {
    int i = blockIdx.x * blockDim.x + threadIdx.x;
    if (i < n) out[i] = (bf16_t)in[i];
}

// transpose (rows x cols) fp32 -> (cols x rows) bf16, so GEMM B-columns are contiguous
__global__ void k_transpose_bf16(const float* __restrict__ in, bf16_t* __restrict__ out,
                                 int rows, int cols) {
    int i = blockIdx.x * blockDim.x + threadIdx.x;
    if (i >= rows * cols) return;
    int r = i / cols, c = i % cols;
    out[(size_t)c * rows + r] = (bf16_t)in[i];
}

// ---------------------------------------------------------------------------
// WMMA bf16 GEMM: C[M,N] = A[M,K] * B^T[N,K] + bias[N]   (fp32 accumulate)
//
// Block = 8 waves = 128x64 output tile. The 64 B-columns x K strip (contiguous
// 32 KB of BT) is staged into LDS once per block via the gfx1250 async
// global->LDS copy (ASYNCcnt), then each wave computes a 16x64 strip: one A
// fragment per K-step reused across 4 v_wmma_f32_16x16x32_bf16 issues, B
// fragments served from LDS.
// Requires: M % 128 == 0, N % 64 == 0, K % 32 == 0, K <= 256.
__global__ void __launch_bounds__(256)
k_gemm_wmma_bf16(const bf16_t* __restrict__ A,
                 const bf16_t* __restrict__ BT,
                 const float*  __restrict__ bias,
                 float* __restrict__ Cout,
                 int M, int N, int K) {
    __shared__ __align__(16) bf16_t lds_bt[64 * 256];   // 32 KB

    int mblocks  = M >> 7;                  // blocks along M (128 rows each)
    int tn_strip = blockIdx.x / mblocks;    // which 64-column strip of N
    int tmb      = blockIdx.x % mblocks;
    int tid  = threadIdx.x;
    int lane = tid & 31;
    int wv   = tid >> 5;

    // ---- stage BT strip (contiguous 64*K bf16) into LDS ----
    {
        const uint4* gsrc = (const uint4*)(BT + (size_t)tn_strip * 64 * K);
        uint4*       ldst = (uint4*)lds_bt;
        int nchunk = (64 * K) >> 3;         // 16-byte chunks
        for (int ch = tid; ch < nchunk; ch += 256) {
#if __has_builtin(__builtin_amdgcn_global_load_async_to_lds_b128)
            __builtin_amdgcn_global_load_async_to_lds_b128(
                (AS1 v4i_t*)(gsrc + ch), (AS3 v4i_t*)(ldst + ch), 0, 0);
#else
            ldst[ch] = gsrc[ch];
#endif
        }
#if __has_builtin(__builtin_amdgcn_global_load_async_to_lds_b128)
#  if __has_builtin(__builtin_amdgcn_s_wait_asynccnt)
        __builtin_amdgcn_s_wait_asynccnt(0);
#  else
        asm volatile("s_wait_asynccnt 0x0" ::: "memory");
#  endif
#endif
        __syncthreads();
    }

    int tm   = tmb * 8 + wv;                // this wave's 16-row tile
    int half = lane >> 4;                   // 0: lanes 0-15, 1: lanes 16-31
    int l    = lane & 15;

    const bf16_t* arow = A + (size_t)(tm * 16 + l) * K;  // A: row M = l (both halves)

    v8f accs[4] = {v8f{}, v8f{}, v8f{}, v8f{}};
    for (int kk = 0; kk < K; kk += 32) {
        __builtin_prefetch(arow + kk + 64, 0, 3);
        // A 16x32 bf16 frag, this code's contiguous k order (B matches): half0 -> K = kk+0..15; half1 -> K = kk+16..31
        v16bf afrag;
        int ak = kk + half * 16;
#pragma unroll
        for (int j = 0; j < 8; ++j) {
            afrag[j]     = arow[ak + j];
            afrag[8 + j] = arow[ak + 8 + j];
        }
        // B 32x16 bf16 frag, same contiguous k order as A: half0 -> K = kk+0..15; half1 -> K = kk+16..31 (col = l)
        int bk = kk + half * 16;
#pragma unroll
        for (int j = 0; j < 4; ++j) {
            const bf16_t* bp = lds_bt + (size_t)(j * 16 + l) * K + bk;
            v16bf bfrag;
#pragma unroll
            for (int q = 0; q < 16; ++q) bfrag[q] = bp[q];
            accs[j] = __builtin_amdgcn_wmma_f32_16x16x32_bf16(
                          false, afrag, false, bfrag, (short)0, accs[j], false, false);
        }
    }
    // D 16x16 f32 layout: element r -> M = r + 8*half, N = l
#pragma unroll
    for (int j = 0; j < 4; ++j) {
        int col = tn_strip * 64 + j * 16 + l;
        float bv = bias ? bias[col] : 0.0f;
#pragma unroll
        for (int r = 0; r < 8; ++r) {
            int row = tm * 16 + r + half * 8;
            Cout[(size_t)row * N + col] = accs[j][r] + bv;
        }
    }
}

// ---------------------------------------------------------------------------
// depthwise 3x3 conv, SAME zero pad, NHWC, 3C channels: y = dw(x) + b
__global__ void k_dwconv(const float* __restrict__ in, const float* __restrict__ w,
                         const float* __restrict__ b, float* __restrict__ out) {
    int i = blockIdx.x * blockDim.x + threadIdx.x;
    if (i >= Bn * Hn * Wn * C3n) return;
    int c  = i % C3n;
    int t  = i / C3n;
    int wx = t % Wn;  t /= Wn;
    int hy = t % Hn;
    int bb = t / Hn;
    float acc = b[c];
#pragma unroll
    for (int ky = 0; ky < 3; ++ky) {
        int yy = hy + ky - 1;
        if (yy < 0 || yy >= Hn) continue;
#pragma unroll
        for (int kx = 0; kx < 3; ++kx) {
            int xx = wx + kx - 1;
            if (xx < 0 || xx >= Wn) continue;
            acc += in[((size_t)(bb * Hn + yy) * Wn + xx) * C3n + c] * w[(ky * 3 + kx) * C3n + c];
        }
    }
    out[i] = acc;
}

// ---------------------------------------------------------------------------
// fold offset head:  eff[g][o][ci*9+t] = sum_co off_w[g][o][co] * off_pconv_w[g][co][ci][t]
__global__ void k_fold_off(const float* __restrict__ pconv,  // (G,CQ,CQ,3,3)
                           const float* __restrict__ lin,    // (G,2K,C)
                           float* __restrict__ eff) {        // (G,2K,CQ*9)
    int i = blockIdx.x * blockDim.x + threadIdx.x;
    int tot = Gn * 2 * Kn * CQn * 9;
    if (i >= tot) return;
    int t  = i % 9;
    int ci = (i / 9) % CQn;
    int o  = (i / (9 * CQn)) % (2 * Kn);
    int g  = i / (9 * CQn * 2 * Kn);
    float acc = 0.f;
    for (int co = 0; co < CQn; ++co)
        acc += lin[((size_t)g * 2 * Kn + o) * Cn + co] *
               pconv[(((size_t)g * CQn + co) * CQn + ci) * 9 + t];
    eff[i] = acc;
}

// fold mask head: eff[o][ci*9+t] = sum_co mask_w[o][co] * mask_pconv_w[co][ci][t]
__global__ void k_fold_mask(const float* __restrict__ pconv, // (CQ,CQ,3,3)
                            const float* __restrict__ lin,   // (K,C)
                            float* __restrict__ eff) {       // (K,CQ*9)
    int i = blockIdx.x * blockDim.x + threadIdx.x;
    int tot = Kn * CQn * 9;
    if (i >= tot) return;
    int t  = i % 9;
    int ci = (i / 9) % CQn;
    int o  = i / (9 * CQn);
    float acc = 0.f;
    for (int co = 0; co < CQn; ++co)
        acc += lin[(size_t)o * Cn + co] * pconv[((size_t)co * CQn + ci) * 9 + t];
    eff[i] = acc;
}

// ---------------------------------------------------------------------------
// offsets: one thread per (b,g,p) -> 2K offset values (folded conv + passthrough + base)
__global__ void k_offsets(const float* __restrict__ qkv,     // (B,H,W,3C), q = channels [0,C)
                          const float* __restrict__ eff,     // (G,2K,CQ*9)
                          const float* __restrict__ lin,     // (G,2K,C)
                          const float* __restrict__ bias,    // (G,2K)
                          float* __restrict__ off_buf) {     // (B,G,P,2K)
    int idx = blockIdx.x * blockDim.x + threadIdx.x;
    if (idx >= Bn * Gn * Pn) return;
    int p = idx % Pn;
    int g = (idx / Pn) % Gn;
    int b = idx / (Pn * Gn);
    int h = p / Wn, w = p % Wn;

    float acc[2 * Kn];
#pragma unroll
    for (int o = 0; o < 2 * Kn; ++o) {
        int kk = o >> 1;
        float base = (o & 1) ? (float)(kk % 3 - 1) : (float)(kk / 3 - 1);
        acc[o] = bias[g * 2 * Kn + o] + base * (float)(2 * g + 1);
    }
    const float* ew = eff + (size_t)g * 2 * Kn * CQn * 9;
    for (int t = 0; t < 9; ++t) {
        int yy = h + t / 3 - 1, xx = w + t % 3 - 1;
        bool inb = (yy >= 0 && yy < Hn && xx >= 0 && xx < Wn);
        const float* pix = qkv + ((size_t)(b * Hn + yy) * Wn + xx) * C3n;
        for (int ci = 0; ci < CQn; ++ci) {
            float qv = inb ? pix[ci] : 0.f;
#pragma unroll
            for (int o = 0; o < 2 * Kn; ++o)
                acc[o] += ew[((size_t)o * CQn + ci) * 9 + t] * qv;
        }
    }
    const float* pix0 = qkv + ((size_t)(b * Hn + h) * Wn + w) * C3n;
    const float* lw   = lin + (size_t)g * 2 * Kn * Cn;
    for (int c = CQn; c < Cn; ++c) {
        float qv = pix0[c];
#pragma unroll
        for (int o = 0; o < 2 * Kn; ++o) acc[o] += lw[(size_t)o * Cn + c] * qv;
    }
    float* dst = off_buf + (size_t)idx * 2 * Kn;
#pragma unroll
    for (int o = 0; o < 2 * Kn; ++o) dst[o] = acc[o];
}

// mask: one thread per (b,p) -> K values
__global__ void k_mask(const float* __restrict__ qkv,
                       const float* __restrict__ eff,     // (K,CQ*9)
                       const float* __restrict__ lin,     // (K,C)
                       const float* __restrict__ bias,    // (K)
                       float* __restrict__ mask_buf) {    // (B,P,K)
    int idx = blockIdx.x * blockDim.x + threadIdx.x;
    if (idx >= Bn * Pn) return;
    int p = idx % Pn, b = idx / Pn;
    int h = p / Wn, w = p % Wn;
    float acc[Kn];
#pragma unroll
    for (int o = 0; o < Kn; ++o) acc[o] = bias[o];
    for (int t = 0; t < 9; ++t) {
        int yy = h + t / 3 - 1, xx = w + t % 3 - 1;
        bool inb = (yy >= 0 && yy < Hn && xx >= 0 && xx < Wn);
        const float* pix = qkv + ((size_t)(b * Hn + yy) * Wn + xx) * C3n;
        for (int ci = 0; ci < CQn; ++ci) {
            float qv = inb ? pix[ci] : 0.f;
#pragma unroll
            for (int o = 0; o < Kn; ++o)
                acc[o] += eff[((size_t)o * CQn + ci) * 9 + t] * qv;
        }
    }
    const float* pix0 = qkv + ((size_t)(b * Hn + h) * Wn + w) * C3n;
    for (int c = CQn; c < Cn; ++c) {
        float qv = pix0[c];
#pragma unroll
        for (int o = 0; o < Kn; ++o) acc[o] += lin[(size_t)o * Cn + c] * qv;
    }
    float* dst = mask_buf + (size_t)idx * Kn;
#pragma unroll
    for (int o = 0; o < Kn; ++o) dst[o] = acc[o];
}

// ---------------------------------------------------------------------------
// deformable attention: one wave per (b,g,p); lane = head channel (HC == 32)
__global__ void k_attn(const float* __restrict__ qkv,      // (B,H,W,3C)
                       const float* __restrict__ off_buf,  // (B,G,P,2K)
                       const float* __restrict__ mask_buf, // (B,P,K)
                       bf16_t* __restrict__ out_bf) {      // (B,P,C) bf16 -> proj GEMM A
    int wid  = (blockIdx.x * blockDim.x + threadIdx.x) >> 5;
    int lane = threadIdx.x & 31;
    if (wid >= Bn * Gn * Pn) return;
    int p = wid % Pn;
    int g = (wid / Pn) % Gn;
    int b = wid / (Pn * Gn);
    int h = p / Wn, w = p % Wn;
    int c = lane;

    const float scale = 0.17677669529663687f;   // 32^-0.5
    const float* pix  = qkv + ((size_t)(b * Hn + h) * Wn + w) * C3n;
    float qv = pix[g * HCn + c] * scale;

    const float* offp = off_buf  + ((size_t)wid) * 2 * Kn;
    const float* mp   = mask_buf + ((size_t)(b * Pn + p)) * Kn;

    int ck = Cn + g * HCn + c;        // k-map channel
    int cv = 2 * Cn + g * HCn + c;    // v-map channel

    float logits[Kn], vvals[Kn];
#pragma unroll
    for (int kk = 0; kk < Kn; ++kk) {
        float cx = fminf(fmaxf((float)w + offp[2 * kk], 0.f), (float)(Wn - 1));
        float cy = fminf(fmaxf((float)h + offp[2 * kk + 1], 0.f), (float)(Hn - 1));
        float x0f = floorf(cx), y0f = floorf(cy);
        int x0 = (int)x0f, y0 = (int)y0f;
        float wx = cx - x0f, wy = cy - y0f;
        int x1 = min(x0 + 1, Wn - 1), y1 = min(y0 + 1, Hn - 1);

        size_t i00 = ((size_t)(b * Hn + y0) * Wn + x0) * C3n;
        size_t i01 = ((size_t)(b * Hn + y0) * Wn + x1) * C3n;
        size_t i10 = ((size_t)(b * Hn + y1) * Wn + x0) * C3n;
        size_t i11 = ((size_t)(b * Hn + y1) * Wn + x1) * C3n;
        float w00 = (1.f - wx) * (1.f - wy), w01 = wx * (1.f - wy);
        float w10 = (1.f - wx) * wy,         w11 = wx * wy;

        float m = mp[kk];
        float kval = (qkv[i00 + ck] * w00 + qkv[i01 + ck] * w01 +
                      qkv[i10 + ck] * w10 + qkv[i11 + ck] * w11) * m;
        float vval = (qkv[i00 + cv] * w00 + qkv[i01 + cv] * w01 +
                      qkv[i10 + cv] * w10 + qkv[i11 + cv] * w11) * m;

        float t = qv * kval;                 // wave-wide dot over 32 head channels
#pragma unroll
        for (int s = 16; s > 0; s >>= 1) t += __shfl_xor(t, s, 32);
        logits[kk] = t;
        vvals[kk]  = vval;
    }
    float mx = logits[0];
#pragma unroll
    for (int kk = 1; kk < Kn; ++kk) mx = fmaxf(mx, logits[kk]);
    float se = 0.f, e[Kn];
#pragma unroll
    for (int kk = 0; kk < Kn; ++kk) { e[kk] = __expf(logits[kk] - mx); se += e[kk]; }
    float inv = 1.f / se, o = 0.f;
#pragma unroll
    for (int kk = 0; kk < Kn; ++kk) o += e[kk] * inv * vvals[kk];

    out_bf[((size_t)(b * Pn + p)) * Cn + g * HCn + c] = (bf16_t)o;
}

// ---------------------------------------------------------------------------
extern "C" void kernel_launch(void* const* d_in, const int* in_sizes, int n_in,
                              void* d_out, int out_size, void* d_ws, size_t ws_size,
                              hipStream_t stream) {
    const float* x           = (const float*)d_in[0];   // (B,H,W,C)
    const float* qkv_w       = (const float*)d_in[1];   // (C,3C)
    const float* qkv_b       = (const float*)d_in[2];   // (3C)
    const float* dw_w        = (const float*)d_in[3];   // (3,3,1,3C)
    const float* dw_b        = (const float*)d_in[4];   // (3C)
    const float* off_pconv_w = (const float*)d_in[5];   // (G,CQ,CQ,3,3)
    const float* off_w       = (const float*)d_in[6];   // (G,2K,C)
    const float* off_b       = (const float*)d_in[7];   // (G,2K)
    const float* mask_pconv_w= (const float*)d_in[8];   // (CQ,CQ,3,3)
    const float* mask_w      = (const float*)d_in[9];   // (K,C)
    const float* mask_b      = (const float*)d_in[10];  // (K)
    const float* proj_w      = (const float*)d_in[11];  // (C,C)
    const float* proj_b      = (const float*)d_in[12];  // (C)
    float* out = (float*)d_out;

    char* ws = (char*)d_ws;
    size_t cur = 0;
    auto sub = [&](size_t bytes) { char* p = ws + cur; cur += (bytes + 255) & ~(size_t)255; return p; };

    bf16_t* xb       = (bf16_t*)sub((size_t)Mtok * Cn * 2);       // x in bf16
    bf16_t* qkv_wT   = (bf16_t*)sub((size_t)C3n * Cn * 2);        // (3C, C) bf16
    bf16_t* proj_wT  = (bf16_t*)sub((size_t)Cn * Cn * 2);         // (C, C)  bf16
    float*  qkv_lin  = (float*) sub((size_t)Mtok * C3n * 4);      // GEMM1 out
    float*  qkv_dw   = (float*) sub((size_t)Mtok * C3n * 4);      // after dwconv
    float*  eff_off  = (float*) sub((size_t)Gn * 2 * Kn * CQn * 9 * 4);
    float*  eff_mask = (float*) sub((size_t)Kn * CQn * 9 * 4);
    float*  off_buf  = (float*) sub((size_t)Bn * Gn * Pn * 2 * Kn * 4);
    float*  mask_buf = (float*) sub((size_t)Bn * Pn * Kn * 4);
    bf16_t* attn_bf  = (bf16_t*)sub((size_t)Mtok * Cn * 2);
    (void)ws_size;

    const int BS = 256;

    // 1) convert / transpose weights to bf16
    k_f32_to_bf16<<<cdiv_i(Mtok * Cn, BS), BS, 0, stream>>>(x, xb, Mtok * Cn);
    k_transpose_bf16<<<cdiv_i(Cn * C3n, BS), BS, 0, stream>>>(qkv_w, qkv_wT, Cn, C3n);
    k_transpose_bf16<<<cdiv_i(Cn * Cn, BS), BS, 0, stream>>>(proj_w, proj_wT, Cn, Cn);

    // 2) QKV GEMM (WMMA, async-LDS-staged B): (8192x256)*(256x768) + b
    k_gemm_wmma_bf16<<<(Mtok / 128) * (C3n / 64), BS, 0, stream>>>(
        xb, qkv_wT, qkv_b, qkv_lin, Mtok, C3n, Cn);

    // 3) depthwise 3x3
    k_dwconv<<<cdiv_i(Mtok * C3n, BS), BS, 0, stream>>>(qkv_lin, dw_w, dw_b, qkv_dw);

    // 4) fold pconv+linear heads into effective weights
    k_fold_off <<<cdiv_i(Gn * 2 * Kn * CQn * 9, BS), BS, 0, stream>>>(off_pconv_w, off_w, eff_off);
    k_fold_mask<<<cdiv_i(Kn * CQn * 9, BS), BS, 0, stream>>>(mask_pconv_w, mask_w, eff_mask);

    // 5) per-pixel offsets and mask logits
    k_offsets<<<cdiv_i(Bn * Gn * Pn, BS), BS, 0, stream>>>(qkv_dw, eff_off, off_w, off_b, off_buf);
    k_mask   <<<cdiv_i(Bn * Pn, BS), BS, 0, stream>>>(qkv_dw, eff_mask, mask_w, mask_b, mask_buf);

    // 6) deformable attention (wave per (b,g,p), lane per head channel)
    k_attn<<<cdiv_i(Bn * Gn * Pn * 32, BS), BS, 0, stream>>>(qkv_dw, off_buf, mask_buf, attn_bf);

    // 7) output projection GEMM (WMMA, async-LDS-staged B): (8192x256)*(256x256) + b -> d_out
    k_gemm_wmma_bf16<<<(Mtok / 128) * (Cn / 64), BS, 0, stream>>>(
        attn_bf, proj_wT, proj_b, out, Mtok, Cn, Cn);

    (void)in_sizes; (void)n_in; (void)out_size;
}
